// JointTransformerBlock_88751204205069
// MI455X (gfx1250) — matrix operand order repaired and checked by reading, not compiled, not run on hardware
//
#include <hip/hip_runtime.h>

// ---------------------------------------------------------------------------
// JointTransformerBlock for MI455X (gfx1250, wave32, WMMA)
//
// Shapes: B=16, N=300(hidden), S=300(source), T=77(text), D=512, H=8, DH=64,
// FF=2048, L=677. Concatenated token order (matches reference output order):
// [text(0..76), source(77..376), hidden(377..676)].
//
// All matmuls use v_wmma_f32_16x16x32_f16 (f16 A/B, f32 accumulate).
// ---------------------------------------------------------------------------

typedef __attribute__((ext_vector_type(16))) _Float16 v16h;
typedef __attribute__((ext_vector_type(8)))  float    v8f;

#define BB   16
#define HH   8
#define DD   512
#define DHH  64
#define TT   77
#define SS   300
#define NNH  300
#define LL   677   // TT + SS + NNH
#define FFD  2048

#define LDA_P 40   // LDS pitch (halves) for 32-wide K tiles (32 + 8 pad)
#define LKV   72   // LDS pitch (halves) for 64-wide tiles   (64 + 8 pad)

// -------------------------------- fragment loaders -------------------------
// A-matrix 16x32 f16: lane m = lane&15;
// halves 0..15 -> K = (lane<16 ? 0 : 16) + h
__device__ inline v16h frag_A(const _Float16* base, int pitch) {
  int lane = threadIdx.x & 31;
  int m = lane & 15;
  int k8 = (lane < 16) ? 0 : 16;
  const _Float16* p = base + m * pitch;
  v16h r{};
#pragma unroll
  for (int h = 0; h < 8; ++h) { r[h] = p[k8 + h]; r[h + 8] = p[8 + k8 + h]; }
  return r;
}

// B-matrix 32x16 f16: lane n = lane&15; halves h -> K = (lane<16?0:16) + h.
// LDS layout is [n][k] so per-lane K is contiguous.
__device__ inline v16h frag_B(const _Float16* base, int pitch) {
  int lane = threadIdx.x & 31;
  int n = lane & 15;
  int kb = (lane < 16) ? 0 : 16;
  const _Float16* p = base + n * pitch + kb;
  v16h r{};
#pragma unroll
  for (int h = 0; h < 16; ++h) r[h] = p[h];
  return r;
}

// ----------------------------- weight convert+transpose --------------------
// in: f32 [K][N] -> out: f16 [N][K]
__global__ void wt_kernel(const float* __restrict__ in, _Float16* __restrict__ out,
                          int K, int N) {
  int idx = blockIdx.x * 256 + threadIdx.x;
  if (idx >= K * N) return;
  int k = idx / N, n = idx % N;
  out[(size_t)n * K + k] = (_Float16)in[idx];
}

// ----------------------------- ada modulation -------------------------------
// ada[s][b][e] = sum_d silu(temb[b][d]) * Wada[s][d][e] + bada[s][e]
__global__ void ada_gemm_kernel(const float* __restrict__ temb,
                                const float* __restrict__ Wada,
                                const float* __restrict__ bada,
                                float* __restrict__ ada) {
  int idx = blockIdx.x * 256 + threadIdx.x;
  if (idx >= 3 * BB * 6 * DD) return;
  int e = idx % (6 * DD);
  int b = (idx / (6 * DD)) % BB;
  int s = idx / (6 * DD * BB);
  float acc = bada[s * 6 * DD + e];
  const float* w  = Wada + (size_t)s * DD * 6 * DD + e;
  const float* tp = temb + b * DD;
  for (int d = 0; d < DD; ++d) {
    float tv = tp[d];
    acc += (tv / (1.f + __expf(-tv))) * w[(size_t)d * 6 * DD];
  }
  ada[idx] = acc;
}

// ------------------------ stream row decomposition --------------------------
__device__ inline void decode_row(int l, int& i, int& local) {
  if (l < TT)            { i = 2; local = l; }
  else if (l < TT + SS)  { i = 1; local = l - TT; }
  else                   { i = 0; local = l - TT - SS; }
}

// --------------------------- LN + AdaLN modulation --------------------------
// n16[row] = LN(stream_row) * (1 + sc_a) + sh_a   (f16), row = b*677 + l
__global__ void __launch_bounds__(256)
norm_mod_kernel(const float* __restrict__ x, const float* __restrict__ src,
                const float* __restrict__ txt, const float* __restrict__ ada,
                _Float16* __restrict__ n16) {
  int row = blockIdx.x;
  int b = row / LL, l = row % LL;
  int i, local; decode_row(l, i, local);
  const float* sp = (i == 2) ? txt + ((size_t)b * TT + local) * DD
                  : (i == 1) ? src + ((size_t)b * SS + local) * DD
                             : x   + ((size_t)b * NNH + local) * DD;
  int t = threadIdx.x;
  float v0 = sp[t], v1 = sp[t + 256];
  float s1 = v0 + v1, s2 = v0 * v0 + v1 * v1;
#pragma unroll
  for (int off = 16; off; off >>= 1) {
    s1 += __shfl_xor(s1, off, 32);
    s2 += __shfl_xor(s2, off, 32);
  }
  __shared__ float ws1[8], ws2[8];
  int wv = t >> 5, ln = t & 31;
  if (ln == 0) { ws1[wv] = s1; ws2[wv] = s2; }
  __syncthreads();
  float ts1 = 0.f, ts2 = 0.f;
#pragma unroll
  for (int w = 0; w < 8; ++w) { ts1 += ws1[w]; ts2 += ws2[w]; }
  float mean = ts1 / (float)DD;
  float var  = ts2 / (float)DD - mean * mean;
  float inv  = rsqrtf(var + 1e-6f);
  const float* ad = ada + ((size_t)i * BB + b) * (6 * DD);
  _Float16* op = n16 + (size_t)row * DD;
  op[t]       = (_Float16)((v0 - mean) * inv * (1.f + ad[DD + t])       + ad[t]);
  op[t + 256] = (_Float16)((v1 - mean) * inv * (1.f + ad[DD + t + 256]) + ad[t + 256]);
}

// ------------------------------- WMMA GEMM ----------------------------------
// C[M,N] = A[M,K] @ WT[N,K]^T + bias, 64x64 tiles, 4 waves x 16 rows.
// A rows map through (len, Apitch, Aoff): arow = (m/len)*Apitch + Aoff + m%len.
// MODE 0: out f32 at orow = (m/len)*Opitch + Ooff + m%len, col n, pitch ldc.
// MODE 1: out f32 qkv-raw [B][H][677][64]; l = Ooff + m%len.
// MODE 2: out f32 = hres[orow,n] + g_m * (acc + bias)   (final residual).
template <int MODE>
__global__ void __launch_bounds__(128)
gemm_kernel(const _Float16* __restrict__ A, const _Float16* __restrict__ WT,
            const float* __restrict__ bias, float* __restrict__ outF,
            int M, int N, int K,
            int len, int Apitch, int Aoff, int Opitch, int Ooff, int ldc,
            const float* __restrict__ hres, const float* __restrict__ adamod,
            int stream_id) {
  __shared__ _Float16 As[64 * LDA_P];
  __shared__ _Float16 Bs[64 * LDA_P];
  const int t = threadIdx.x, wave = t >> 5, lane = t & 31;
  const int tileN = blockIdx.x * 64;
  const int tileM = blockIdx.y * 64;

  v8f acc[4] = {};

  for (int k0 = 0; k0 < K; k0 += 32) {
    { // stage A tile (64 x 32), zero-padded past M
      int row = t >> 1, seg = (t & 1) * 16;
      _Float16* dst = &As[row * LDA_P + seg];
      int m = tileM + row;
      if (m < M) {
        int arow = (m / len) * Apitch + Aoff + (m % len);
        const _Float16* sp = A + (size_t)arow * K + k0 + seg;
        if (k0 + 32 < K) __builtin_prefetch(sp + 32, 0, 1);
#pragma unroll
        for (int q = 0; q < 16; ++q) dst[q] = sp[q];
      } else {
#pragma unroll
        for (int q = 0; q < 16; ++q) dst[q] = (_Float16)0.f;
      }
    }
    { // stage B tile as [n][k] (64 x 32) from WT [N][K]
      int row = t >> 1, seg = (t & 1) * 16;
      _Float16* dst = &Bs[row * LDA_P + seg];
      int n = tileN + row;
      if (n < N) {
        const _Float16* sp = WT + (size_t)n * K + k0 + seg;
        if (k0 + 32 < K) __builtin_prefetch(sp + 32, 0, 1);
#pragma unroll
        for (int q = 0; q < 16; ++q) dst[q] = sp[q];
      } else {
#pragma unroll
        for (int q = 0; q < 16; ++q) dst[q] = (_Float16)0.f;
      }
    }
    __syncthreads();

    v16h a = frag_A(&As[(wave * 16) * LDA_P], LDA_P);
#pragma unroll
    for (int nt = 0; nt < 4; ++nt) {
      v16h bf = frag_B(&Bs[(nt * 16) * LDA_P], LDA_P);
      acc[nt] = __builtin_amdgcn_wmma_f32_16x16x32_f16(
          false, a, false, bf, (short)0, acc[nt], false, false);
    }
    __syncthreads();
  }

  // epilogue (C layout: row = r + 8*(lane>=16), col = lane&15)
  const int m0 = tileM + wave * 16 + ((lane >= 16) ? 8 : 0);
  const int nc = tileN + (lane & 15);
#pragma unroll
  for (int nt = 0; nt < 4; ++nt) {
    int n = nc + nt * 16;
    if (n >= N) continue;
    float bn = bias ? bias[n] : 0.f;
#pragma unroll
    for (int r = 0; r < 8; ++r) {
      int m = m0 + r;
      if (m >= M) continue;
      float v = acc[nt][r] + bn;
      int bidx = m / len, local = m % len;
      if (MODE == 0) {
        int orow = bidx * Opitch + Ooff + local;
        outF[(size_t)orow * ldc + n] = v;
      } else if (MODE == 1) {
        int l = Ooff + local;
        outF[((((size_t)bidx * HH) + (n >> 6)) * LL + l) * DHH + (n & 63)] = v;
      } else {
        int orow = bidx * Opitch + Ooff + local;
        float g = adamod[((size_t)stream_id * BB + bidx) * (6 * DD) + 5 * DD + n];
        outF[(size_t)orow * ldc + n] = hres[(size_t)orow * ldc + n] + g * v;
      }
    }
  }
}

// ------------------- QKV epilogue: RMS-norm + xpos-RoPE / pos ---------------
// raw f32 [B][H][L][DH] -> f16 same layout.  kind: 0=q, 1=k, 2=v.
__global__ void __launch_bounds__(64)
qkv_post_kernel(const float* __restrict__ raw, const float* __restrict__ gqk,
                const float* __restrict__ pos, _Float16* __restrict__ out16,
                int kind) {
  int row = blockIdx.x;             // (b*H + h)*L + l
  int l = row % LL;
  int d = threadIdx.x;
  float v = raw[(size_t)row * DHH + d];
  if (kind == 2) { out16[(size_t)row * DHH + d] = (_Float16)v; return; }

  int i, tloc; decode_row(l, i, tloc);
  int Lq = (i == 2) ? TT : ((i == 1) ? SS : NNH);

  __shared__ float red[64];
  red[d] = v * v;
  __syncthreads();
#pragma unroll
  for (int s2 = 32; s2 > 0; s2 >>= 1) {
    if (d < s2) red[d] += red[d + s2];
    __syncthreads();
  }
  float ms = red[0] / (float)DHH;
  float g = gqk[((size_t)i * 2 + kind) * DHH + d];
  v = v * rsqrtf(ms + 1e-6f) * g;

  if (i == 2) {
    v += pos[(size_t)tloc * DHH + d];       // learned pos-embed (text)
  } else {
    int p = d >> 1;
    float inv   = __powf(10000.f, -(float)(2 * p) / (float)DHH);
    float ang   = (float)tloc * inv;
    float basep = (2.f * p + 0.4f * DHH) / (1.4f * DHH);
    float power = ((float)tloc - (float)(Lq / 2)) / 512.f;
    float scale = __powf(basep, power);
    if (kind == 1) scale = 1.f / scale;
    float c = __cosf(ang), sn = __sinf(ang);
    float other = __shfl_xor(v, 1, 32);
    float rot = (d & 1) ? other : -other;   // rotate_half
    v = v * (c * scale) + rot * (sn * scale);
  }
  out16[(size_t)row * DHH + d] = (_Float16)v;
}

// ----------------------------- flash attention ------------------------------
// grid: (ceil(L/64), B*H), block 128 (4 waves, 16 q-rows each).
__global__ void __launch_bounds__(128)
attn_kernel(const _Float16* __restrict__ Q, const _Float16* __restrict__ Kk,
            const _Float16* __restrict__ V, const int* __restrict__ valid,
            _Float16* __restrict__ O) {
  __shared__ _Float16 Ks[64 * LKV];          // [key][dh]
  __shared__ _Float16 Vs[64 * LKV];          // [dh][key]  (transposed)
  __shared__ _Float16 Ps[4 * 16 * LKV];      // per-wave P tile [qrow][key]
  __shared__ unsigned char Sv[704];

  const int t = threadIdx.x, wave = t >> 5, lane = t & 31;
  const int bh = blockIdx.y, b = bh >> 3, hh = bh & 7;
  const int iqw = blockIdx.x * 64 + wave * 16;
  const size_t base = (size_t)bh * LL * DHH;

  for (int l = t; l < 704; l += 128)
    Sv[l] = (l < LL) ? (unsigned char)(valid[b * LL + l] != 0) : 0;
  __syncthreads();

  // Q fragments (K-dim = DH = 64 -> two 16x32 A chunks), kept in registers.
  const int k8 = (lane < 16) ? 0 : 16;
  const int mq = iqw + (lane & 15);
  v16h a0{}, a1{};
  if (mq < LL) {
    const _Float16* qp = Q + base + (size_t)mq * DHH;
#pragma unroll
    for (int h = 0; h < 8; ++h) {
      a0[h] = qp[k8 + h];      a0[h + 8] = qp[8 + k8 + h];
      a1[h] = qp[32 + k8 + h]; a1[h + 8] = qp[40 + k8 + h];
    }
  }

  const int mrow = (lane >= 16) ? 8 : 0;
  bool vq[8]; bool iqok[8];
#pragma unroll
  for (int r = 0; r < 8; ++r) {
    int iq = iqw + mrow + r;
    iqok[r] = iq < LL;
    vq[r] = iqok[r] && Sv[iq];
  }

  const float MASKV = -3.0e38f;
  float mrun[8], lrun[8];
  v8f acc[4] = {};
#pragma unroll
  for (int r = 0; r < 8; ++r) { mrun[r] = -1e30f; lrun[r] = 0.f; }

  const int numKV = (LL + 63) >> 6;
  for (int jt = 0; jt < numKV; ++jt) {
    const int j0 = jt * 64;
    { // stage K tile [key][dh]
      int key = t >> 1, seg = (t & 1) * 32;
      _Float16* dst = &Ks[key * LKV + seg];
      int jk = j0 + key;
      if (jk < LL) {
        const _Float16* sp = Kk + base + (size_t)jk * DHH + seg;
#pragma unroll
        for (int q = 0; q < 32; ++q) dst[q] = sp[q];
      } else {
#pragma unroll
        for (int q = 0; q < 32; ++q) dst[q] = (_Float16)0.f;
      }
    }
    { // stage V tile transposed [dh][key]
      int key = t >> 1, seg = (t & 1) * 32;
      int jk = j0 + key;
      if (jk < LL) {
        const _Float16* sp = V + base + (size_t)jk * DHH + seg;
#pragma unroll
        for (int q = 0; q < 32; ++q) Vs[(seg + q) * LKV + key] = sp[q];
      } else {
#pragma unroll
        for (int q = 0; q < 32; ++q) Vs[(seg + q) * LKV + key] = (_Float16)0.f;
      }
    }
    __syncthreads();

    // scores S = Q @ K^T  (16x64 per wave, 4 sub-tiles x 2 chained WMMAs)
    float pv[4][8];
#pragma unroll
    for (int nt = 0; nt < 4; ++nt) {
      int n = nt * 16 + (lane & 15);
      int kb = (lane < 16) ? 0 : 16;
      const _Float16* kp = &Ks[n * LKV];
      v16h b0{}, b1{};
#pragma unroll
      for (int h = 0; h < 16; ++h) { b0[h] = kp[kb + h]; b1[h] = kp[32 + kb + h]; }
      v8f c = {};
      c = __builtin_amdgcn_wmma_f32_16x16x32_f16(false, a0, false, b0, (short)0, c, false, false);
      c = __builtin_amdgcn_wmma_f32_16x16x32_f16(false, a1, false, b1, (short)0, c, false, false);
      int jk = j0 + n;
      bool kin = jk < LL;
      bool vk = kin && Sv[jk];
#pragma unroll
      for (int r = 0; r < 8; ++r) {
        int iq = iqw + mrow + r;
        bool ok = kin && iqok[r] && ((vk && vq[r]) || (iq == jk));
        pv[nt][r] = ok ? c[r] * 0.125f : MASKV;   // 1/sqrt(64)
      }
    }

    // online softmax (rows live across 16-lane halves)
#pragma unroll
    for (int r = 0; r < 8; ++r) {
      float lm = fmaxf(fmaxf(pv[0][r], pv[1][r]), fmaxf(pv[2][r], pv[3][r]));
#pragma unroll
      for (int off = 1; off < 16; off <<= 1) lm = fmaxf(lm, __shfl_xor(lm, off, 32));
      float mnew = fmaxf(mrun[r], lm);
      float sc = __expf(mrun[r] - mnew);
      mrun[r] = mnew;
      float rs = 0.f;
#pragma unroll
      for (int nt = 0; nt < 4; ++nt) {
        float e = __expf(pv[nt][r] - mnew);
        pv[nt][r] = e;
        rs += e;
      }
#pragma unroll
      for (int off = 1; off < 16; off <<= 1) rs += __shfl_xor(rs, off, 32);
      lrun[r] = lrun[r] * sc + rs;
#pragma unroll
      for (int nt = 0; nt < 4; ++nt) acc[nt][r] *= sc;
    }

    // restage P through LDS (C-layout -> A-layout)
    _Float16* pw = &Ps[wave * 16 * LKV];
#pragma unroll
    for (int nt = 0; nt < 4; ++nt) {
      int jl = nt * 16 + (lane & 15);
#pragma unroll
      for (int r = 0; r < 8; ++r)
        pw[(mrow + r) * LKV + jl] = (_Float16)pv[nt][r];
    }
    __syncthreads();

    v16h p0{}, p1{};
    {
      const _Float16* pp = &Ps[wave * 16 * LKV + (lane & 15) * LKV];
#pragma unroll
      for (int h = 0; h < 8; ++h) {
        p0[h] = pp[k8 + h];      p0[h + 8] = pp[8 + k8 + h];
        p1[h] = pp[32 + k8 + h]; p1[h + 8] = pp[40 + k8 + h];
      }
    }
    // O += P @ V  (V as B-fragment from transposed Vs: contiguous key per lane)
#pragma unroll
    for (int nt = 0; nt < 4; ++nt) {
      int n = nt * 16 + (lane & 15);   // dh
      int kb = (lane < 16) ? 0 : 16;
      const _Float16* vp = &Vs[n * LKV];
      v16h b0{}, b1{};
#pragma unroll
      for (int h = 0; h < 16; ++h) { b0[h] = vp[kb + h]; b1[h] = vp[32 + kb + h]; }
      acc[nt] = __builtin_amdgcn_wmma_f32_16x16x32_f16(false, p0, false, b0, (short)0, acc[nt], false, false);
      acc[nt] = __builtin_amdgcn_wmma_f32_16x16x32_f16(false, p1, false, b1, (short)0, acc[nt], false, false);
    }
    __syncthreads();
  }

  // write O as [B][L][H*DH] f16 (A-matrix input for O-projection GEMM)
#pragma unroll
  for (int nt = 0; nt < 4; ++nt) {
    int dh = nt * 16 + (lane & 15);
#pragma unroll
    for (int r = 0; r < 8; ++r) {
      int iq = iqw + mrow + r;
      if (iq < LL)
        O[((size_t)b * LL + iq) * DD + hh * DHH + dh] = (_Float16)(acc[nt][r] / lrun[r]);
    }
  }
}

// --------------------- residual + gate + LN2 + modulation -------------------
__global__ void __launch_bounds__(256)
resid_norm2_kernel(const float* __restrict__ x, const float* __restrict__ src,
                   const float* __restrict__ txt, const float* __restrict__ tmp,
                   const float* __restrict__ ada,
                   float* __restrict__ hout, _Float16* __restrict__ n2out) {
  int row = blockIdx.x;
  int b = row / LL, l = row % LL;
  int i, local; decode_row(l, i, local);
  const float* sp = (i == 2) ? txt + ((size_t)b * TT + local) * DD
                  : (i == 1) ? src + ((size_t)b * SS + local) * DD
                             : x   + ((size_t)b * NNH + local) * DD;
  int t = threadIdx.x;
  const float* ad = ada + ((size_t)i * BB + b) * (6 * DD);
  const float* tp = tmp + (size_t)row * DD;
  float h0 = sp[t]       + ad[2 * DD + t]       * tp[t];
  float h1 = sp[t + 256] + ad[2 * DD + t + 256] * tp[t + 256];
  float* hp = hout + (size_t)row * DD;
  hp[t] = h0; hp[t + 256] = h1;

  float s1 = h0 + h1, s2 = h0 * h0 + h1 * h1;
#pragma unroll
  for (int off = 16; off; off >>= 1) {
    s1 += __shfl_xor(s1, off, 32);
    s2 += __shfl_xor(s2, off, 32);
  }
  __shared__ float ws1[8], ws2[8];
  int wv = t >> 5, ln = t & 31;
  if (ln == 0) { ws1[wv] = s1; ws2[wv] = s2; }
  __syncthreads();
  float ts1 = 0.f, ts2 = 0.f;
#pragma unroll
  for (int w = 0; w < 8; ++w) { ts1 += ws1[w]; ts2 += ws2[w]; }
  float mean = ts1 / (float)DD;
  float var  = ts2 / (float)DD - mean * mean;
  float inv  = rsqrtf(var + 1e-6f);
  _Float16* op = n2out + (size_t)row * DD;
  op[t]       = (_Float16)((h0 - mean) * inv * (1.f + ad[4 * DD + t])       + ad[3 * DD + t]);
  op[t + 256] = (_Float16)((h1 - mean) * inv * (1.f + ad[4 * DD + t + 256]) + ad[3 * DD + t + 256]);
}

// ------------------------------- GLU (exact GELU) ---------------------------
__global__ void glu_kernel(const float* __restrict__ u, _Float16* __restrict__ out,
                           int Mi) {
  int idx = blockIdx.x * 256 + threadIdx.x;
  if (idx >= Mi * FFD) return;
  int m = idx / FFD, c = idx % FFD;
  float lin = u[(size_t)m * (2 * FFD) + c];
  float g   = u[(size_t)m * (2 * FFD) + FFD + c];
  float ge  = 0.5f * g * (1.f + erff(g * 0.70710678118654752f));
  out[idx] = (_Float16)(lin * ge);
}

// =============================== host side ==================================
extern "C" void kernel_launch(void* const* d_in, const int* in_sizes, int n_in,
                              void* d_out, int out_size, void* d_ws, size_t ws_size,
                              hipStream_t stream) {
  (void)in_sizes; (void)n_in; (void)out_size; (void)ws_size;
  const float* x     = (const float*)d_in[0];
  const float* src   = (const float*)d_in[1];
  const float* txt   = (const float*)d_in[2];
  const float* temb  = (const float*)d_in[3];
  const float* Wqkv  = (const float*)d_in[4];
  const float* bqkv  = (const float*)d_in[5];
  const float* Wo    = (const float*)d_in[6];
  const float* bo    = (const float*)d_in[7];
  const float* gqk   = (const float*)d_in[8];
  const float* Wada  = (const float*)d_in[9];
  const float* bada  = (const float*)d_in[10];
  const float* Wff1  = (const float*)d_in[11];
  const float* bff1  = (const float*)d_in[12];
  const float* Wff2  = (const float*)d_in[13];
  const float* bff2  = (const float*)d_in[14];
  const float* pos_q = (const float*)d_in[15];
  const float* pos_k = (const float*)d_in[16];
  const int*   valid = (const int*)d_in[17];
  float* out = (float*)d_out;

  // ----- workspace layout -----
  char* ws = (char*)d_ws;
  size_t off = 0;
  auto alloc = [&](size_t bytes) {
    size_t o = off; off = (off + bytes + 255) & ~(size_t)255; return o;
  };
  const size_t NROW = (size_t)BB * LL;                 // 10832
  const size_t QKVE = (size_t)BB * HH * LL * DHH;      // 5,545,984
  _Float16* wqkv16 = (_Float16*)(ws + alloc(9u * DD * DD * sizeof(_Float16)));
  _Float16* wo16   = (_Float16*)(ws + alloc(3u * DD * DD * sizeof(_Float16)));
  _Float16* wff1_16= (_Float16*)(ws + alloc(3u * (size_t)DD * 2 * FFD * sizeof(_Float16)));
  _Float16* wff2_16= (_Float16*)(ws + alloc(3u * (size_t)FFD * DD * sizeof(_Float16)));
  float*    adaW   = (float*)   (ws + alloc(3u * BB * 6 * DD * sizeof(float)));
  _Float16* n16    = (_Float16*)(ws + alloc(NROW * DD * sizeof(_Float16)));
  float*    qraw   = (float*)   (ws + alloc(QKVE * sizeof(float)));
  float*    kraw   = (float*)   (ws + alloc(QKVE * sizeof(float)));
  float*    vraw   = (float*)   (ws + alloc(QKVE * sizeof(float)));
  _Float16* qf16   = (_Float16*)(ws + alloc(QKVE * sizeof(_Float16)));
  _Float16* kf16   = (_Float16*)(ws + alloc(QKVE * sizeof(_Float16)));
  _Float16* vf16   = (_Float16*)(ws + alloc(QKVE * sizeof(_Float16)));
  _Float16* o16    = (_Float16*)(ws + alloc(NROW * DD * sizeof(_Float16)));
  float*    tmp    = (float*)   (ws + alloc(NROW * DD * sizeof(float)));
  float*    hbuf   = (float*)   (ws + alloc(NROW * DD * sizeof(float)));
  _Float16* n2_16  = (_Float16*)(ws + alloc(NROW * DD * sizeof(_Float16)));
  float*    ubuf   = (float*)   (ws + alloc((size_t)BB * SS * 2 * FFD * sizeof(float)));
  _Float16* ffin   = (_Float16*)(ws + alloc((size_t)BB * SS * FFD * sizeof(_Float16)));

  const int offL[3] = {TT + SS, TT, 0};   // stream i -> offset in L (0=hidden)
  const int lenL[3] = {NNH, SS, TT};

  // 1) weight convert + transpose -> f16 [N][K]
  auto launch_wt = [&](const float* in, _Float16* o, int K, int N) {
    int tot = K * N;
    wt_kernel<<<(tot + 255) / 256, 256, 0, stream>>>(in, o, K, N);
  };
  for (int s = 0; s < 9; ++s)
    launch_wt(Wqkv + (size_t)s * DD * DD, wqkv16 + (size_t)s * DD * DD, DD, DD);
  for (int s = 0; s < 3; ++s)
    launch_wt(Wo + (size_t)s * DD * DD, wo16 + (size_t)s * DD * DD, DD, DD);
  for (int s = 0; s < 3; ++s)
    launch_wt(Wff1 + (size_t)s * DD * 2 * FFD, wff1_16 + (size_t)s * DD * 2 * FFD, DD, 2 * FFD);
  for (int s = 0; s < 3; ++s)
    launch_wt(Wff2 + (size_t)s * FFD * DD, wff2_16 + (size_t)s * FFD * DD, FFD, DD);

  // 2) AdaLN modulation vectors
  {
    int tot = 3 * BB * 6 * DD;
    ada_gemm_kernel<<<(tot + 255) / 256, 256, 0, stream>>>(temb, Wada, bada, adaW);
  }

  // 3) LN + modulation -> n16 (f16, concat layout)
  norm_mod_kernel<<<(int)NROW, 256, 0, stream>>>(x, src, txt, adaW, n16);

  // 4) QKV projections (WMMA) -> raw f32 [B][H][L][DH]
  for (int i = 0; i < 3; ++i) {
    int Mi = BB * lenL[i];
    dim3 g(DD / 64, (Mi + 63) / 64);
    for (int j = 0; j < 3; ++j) {
      float* dst = (j == 0) ? qraw : (j == 1) ? kraw : vraw;
      gemm_kernel<1><<<g, 128, 0, stream>>>(
          n16, wqkv16 + (size_t)(i * 3 + j) * DD * DD, bqkv + (i * 3 + j) * DD,
          dst, Mi, DD, DD, lenL[i], LL, offL[i], 0, offL[i], 0,
          nullptr, nullptr, 0);
    }
  }

  // 5) RMS-norm + RoPE(xpos)/pos-embed, f32 -> f16
  {
    int blocks = BB * HH * LL;
    qkv_post_kernel<<<blocks, 64, 0, stream>>>(qraw, gqk, pos_q, qf16, 0);
    qkv_post_kernel<<<blocks, 64, 0, stream>>>(kraw, gqk, pos_k, kf16, 1);
    qkv_post_kernel<<<blocks, 64, 0, stream>>>(vraw, gqk, nullptr, vf16, 2);
  }

  // 6) flash attention (WMMA) -> o16 [B][L][H*DH]
  {
    dim3 g((LL + 63) / 64, BB * HH);
    attn_kernel<<<g, 128, 0, stream>>>(qf16, kf16, vf16, valid, o16);
  }

  // 7) output projection (WMMA) -> tmp f32 (concat layout)
  for (int i = 0; i < 3; ++i) {
    int Mi = BB * lenL[i];
    dim3 g(DD / 64, (Mi + 63) / 64);
    gemm_kernel<0><<<g, 128, 0, stream>>>(
        o16, wo16 + (size_t)i * DD * DD, bo + i * DD, tmp,
        Mi, DD, DD, lenL[i], LL, offL[i], LL, offL[i], DD,
        nullptr, nullptr, 0);
  }

  // 8) h = s + g_a*attn ; n2 = LN(h)*(1+sc_m)+sh_m
  resid_norm2_kernel<<<(int)NROW, 256, 0, stream>>>(x, src, txt, tmp, adaW, hbuf, n2_16);

  // 9) FF per stream: ff1 (WMMA) -> GLU-GELU -> ff2 (WMMA, fused final residual)
  for (int i = 0; i < 3; ++i) {
    int Mi = BB * lenL[i];
    {
      dim3 g((2 * FFD) / 64, (Mi + 63) / 64);
      gemm_kernel<0><<<g, 128, 0, stream>>>(
          n2_16, wff1_16 + (size_t)i * DD * 2 * FFD, bff1 + i * 2 * FFD, ubuf,
          Mi, 2 * FFD, DD, lenL[i], LL, offL[i], lenL[i], 0, 2 * FFD,
          nullptr, nullptr, 0);
    }
    glu_kernel<<<(Mi * FFD + 255) / 256, 256, 0, stream>>>(ubuf, ffin, Mi);
    {
      dim3 g(DD / 64, (Mi + 63) / 64);
      gemm_kernel<2><<<g, 128, 0, stream>>>(
          ffin, wff2_16 + (size_t)i * FFD * DD, bff2 + i * DD, out,
          Mi, DD, FFD, lenL[i], lenL[i], 0, LL, offL[i], DD,
          hbuf, adaW, i);
    }
  }
}
